// FP4Linear_89060441850294
// MI455X (gfx1250) — matrix operand order repaired and checked by reading, not compiled, not run on hardware
//
#include <hip/hip_runtime.h>

// ---------------------------------------------------------------------------
// FP4 (E2M1 + sign) block-quantized linear:  out = x @ W^T + bias
//   x      : [8192, 4096]  fp32
//   packed : [33554432]    int32, each holding ONE byte = two fp4 codes
//                          (element 2i = high nibble, 2i+1 = low nibble)
//   scales : [524288]      fp32, one per 128 contiguous weights (along K)
//   bias   : [16384]       fp32
//   out    : [8192, 16384] fp32
//
// Compute-bound GEMM (AI ~1.4 kFLOP/byte >> roofline knee): dequantize W to
// f16 on the fly (branchless, co-executes with XDL), convert x to f16 on the
// fly, accumulate f32 via v_wmma_f32_16x16x32_f16.
// Block tile 128(M) x 256(N), 8 wave32 waves, each wave = 64x64 = 4x4 WMMA
// tiles -> 16 WMMAs per K-step per wave vs 16 fragment ds_load_b128.
// ---------------------------------------------------------------------------

typedef _Float16 v4h  __attribute__((ext_vector_type(4)));
typedef _Float16 v8h  __attribute__((ext_vector_type(8)));
typedef _Float16 v16h __attribute__((ext_vector_type(16)));
typedef float    v8f  __attribute__((ext_vector_type(8)));

constexpr int OUT_F  = 16384;
constexpr int IN_F   = 4096;
constexpr int TOKENS = 8192;

constexpr int BM = 128;          // M tile (tokens)
constexpr int BN = 256;          // N tile (out features)
constexpr int BK = 32;           // K step  (= WMMA f16 K)
constexpr int LDT = BK + 8;      // LDS row stride in halfs (80B, 16B-aligned)

// Branchless FP4 E2M1 -> f16 * scale.
// Magnitudes {0,.5,1,1.5,2,3,4,6} = {0,1,2,3,4,6,8,12} * 0.5; the integer
// units all fit a nibble -> one 32-bit LUT constant, no selects/branches.
// hsc = 0.5f * scale; sign applied by XOR on the f32 sign bit.
__device__ __forceinline__ _Float16 dq_fp4(unsigned c, float hsc) {
    unsigned u  = (0xC8643210u >> ((c & 7u) * 4u)) & 0xFu;   // mag / 0.5
    float    v  = (float)u * hsc;
    unsigned vb = __float_as_uint(v) ^ ((c & 8u) << 28);     // sign bit
    return (_Float16)__uint_as_float(vb);
}

__device__ __forceinline__ v16h cat8(v8h lo, v8h hi) {
    return __builtin_shufflevector(lo, hi, 0,1,2,3,4,5,6,7,8,9,10,11,12,13,14,15);
}

__global__ __launch_bounds__(256)
void fp4_linear_wmma(const float* __restrict__ x,
                     const int*   __restrict__ packed,
                     const float* __restrict__ scales,
                     const float* __restrict__ bias,
                     float*       __restrict__ out)
{
    __shared__ _Float16 Asmem[BM][LDT];   // x tile, f16, row-major [M][K]
    __shared__ _Float16 Bsmem[BN][LDT];   // W tile, f16, row-major [N][K]

    const int tid   = threadIdx.x;
    const int lane  = tid & 31;
    const int wave  = tid >> 5;       // 0..7
    const int waveM = wave >> 2;      // 0..1  -> 64 rows of M each
    const int waveN = wave & 3;       // 0..3  -> 64 cols of N each
    const int r     = lane & 15;      // row/col within 16
    const int g     = lane >> 4;      // lane half select

    const int m0 = blockIdx.y * BM;
    const int n0 = blockIdx.x * BN;

    // 4 (M) x 4 (N) accumulator tiles of 16x16 f32 -> 128 VGPRs
    v8f acc[4][4];
#pragma unroll
    for (int i = 0; i < 4; ++i)
#pragma unroll
        for (int j = 0; j < 4; ++j)
#pragma unroll
            for (int e = 0; e < 8; ++e) acc[i][j][e] = 0.0f;

    // global-load work split (256 threads):
    //   x tile 128 x 32 fp32 : 4x float4 per thread
    const int xRow = tid >> 3;            // 0..31  (+32*i)
    const int xCol = (tid & 7) * 4;       // 0..28
    //   packed tile 256 rows x 16 ints : one full row (4x int4) per thread
    const int nGlob = n0 + tid;

    const float* xBase = x + (size_t)m0 * IN_F;
    const int*   pBase = packed + (size_t)nGlob * (IN_F / 2);
    const float* sBase = scales + (size_t)nGlob * (IN_F / 128);

    for (int kk = 0; kk < IN_F / BK; ++kk) {
        const int k0 = kk * BK;

        // ------- global loads (hoisted above barrier to hide latency) -------
        float4 xf[4];
#pragma unroll
        for (int i = 0; i < 4; ++i)
            xf[i] = *(const float4*)(xBase + (size_t)(xRow + 32 * i) * IN_F + k0 + xCol);

        const int* pc = pBase + (k0 >> 1);
        int4 pb[4];
#pragma unroll
        for (int i = 0; i < 4; ++i) pb[i] = *(const int4*)(pc + 4 * i);
        // one scale per (row, 128-K-block); a 32-wide K step never straddles a block
        const float hsc = 0.5f * sBase[k0 >> 7];

        if (kk + 1 < IN_F / BK) {   // prefetch next K slice -> global_prefetch_b8
            __builtin_prefetch(xBase + (size_t)xRow * IN_F + k0 + BK + xCol, 0, 0);
            __builtin_prefetch(pc + (BK >> 1), 0, 0);
        }

        __syncthreads();   // previous iteration's LDS reads complete

        // ------- x tile -> f16 LDS (8B vector store) -------
#pragma unroll
        for (int i = 0; i < 4; ++i) {
            v4h h;
            h[0] = (_Float16)xf[i].x;
            h[1] = (_Float16)xf[i].y;
            h[2] = (_Float16)xf[i].z;
            h[3] = (_Float16)xf[i].w;
            *(v4h*)&Asmem[xRow + 32 * i][xCol] = h;
        }

        // ------- dequantize 32 fp4 codes -> f16 LDS (4x 16B vector stores) ---
        {
            _Float16 tmp[32];
            const int by[16] = {pb[0].x, pb[0].y, pb[0].z, pb[0].w,
                                pb[1].x, pb[1].y, pb[1].z, pb[1].w,
                                pb[2].x, pb[2].y, pb[2].z, pb[2].w,
                                pb[3].x, pb[3].y, pb[3].z, pb[3].w};
#pragma unroll
            for (int j = 0; j < 16; ++j) {
                unsigned b = (unsigned)by[j] & 0xFFu;
                tmp[2 * j]     = dq_fp4(b >> 4, hsc);    // element 2j   = high nibble
                tmp[2 * j + 1] = dq_fp4(b & 0xFu, hsc);  // element 2j+1 = low nibble
            }
#pragma unroll
            for (int j = 0; j < 4; ++j)
                *(v8h*)&Bsmem[tid][8 * j] = *(const v8h*)&tmp[8 * j];
        }

        __syncthreads();

        // ------- fragment loads + WMMA -------
        // A (16-bit 16x32, ISA 7.12.2): lane half g: halves 0..7 = K g*8..g*8+7,
        //                               halves 8..15 = K 16+g*8 .. 16+g*8+7
        v16h afrag[4];
#pragma unroll
        for (int tm = 0; tm < 4; ++tm) {
            const _Float16* arow = Asmem[waveM * 64 + tm * 16 + r];
            afrag[tm] = cat8(*(const v8h*)(arow + g * 8),
                             *(const v8h*)(arow + 16 + g * 8));
        }

#pragma unroll
        for (int tn = 0; tn < 4; ++tn) {
            // B (32x16): lane = N column, same K order as A: halves 0..7 = K g*8..g*8+7,
            // halves 8..15 = K 16+g*8 .. 16+g*8+7
            const _Float16* brow = Bsmem[waveN * 64 + tn * 16 + r];
            v16h bfrag = cat8(*(const v8h*)(brow + g * 8),
                              *(const v8h*)(brow + 16 + g * 8));
#pragma unroll
            for (int tm = 0; tm < 4; ++tm) {
                acc[tm][tn] = __builtin_amdgcn_wmma_f32_16x16x32_f16(
                    false, afrag[tm], false, bfrag,
                    (short)0, acc[tm][tn], false, false);
            }
        }
    }

    // ------- epilogue: bias + store (C/D layout: VGPR v -> M = v + 8*g, N = r) -------
#pragma unroll
    for (int tn = 0; tn < 4; ++tn) {
        const int nc = n0 + waveN * 64 + tn * 16 + r;
        const float bv = bias[nc];
#pragma unroll
        for (int tm = 0; tm < 4; ++tm) {
            const int mBase = m0 + waveM * 64 + tm * 16 + 8 * g;
#pragma unroll
            for (int v = 0; v < 8; ++v)
                out[(size_t)(mBase + v) * OUT_F + nc] = acc[tm][tn][v] + bv;
        }
    }
}

extern "C" void kernel_launch(void* const* d_in, const int* in_sizes, int n_in,
                              void* d_out, int out_size, void* d_ws, size_t ws_size,
                              hipStream_t stream) {
    (void)in_sizes; (void)n_in; (void)d_ws; (void)ws_size; (void)out_size;
    const float* x      = (const float*)d_in[0];
    const int*   packed = (const int*)d_in[1];
    const float* scales = (const float*)d_in[2];
    const float* bias   = (const float*)d_in[3];
    float*       out    = (float*)d_out;

    dim3 grid(OUT_F / BN, TOKENS / BM);   // 64 x 64 workgroups
    fp4_linear_wmma<<<grid, 256, 0, stream>>>(x, packed, scales, bias, out);
}
